// Attention_18485539242642
// MI455X (gfx1250) — matrix operand order repaired and checked by reading, not compiled, not run on hardware
//
#include <hip/hip_runtime.h>
#include <math.h>

typedef _Float16 f16;
typedef __attribute__((ext_vector_type(8)))  _Float16 v8h;
typedef __attribute__((ext_vector_type(16))) _Float16 v16h;
typedef __attribute__((ext_vector_type(8)))  float    v8f;

#define LQ 1024
#define NBATCH 8
#define CD 1024
#define NH 16
#define HD 64
#define MROWS (LQ * NBATCH) /* 8192 */

// ---- fragment loaders (CDNA5 wave32 WMMA 16x16x32 f16 layouts) ----
// A (16x32, f16): lane m (0-15) VGPR0-3 = K[8g..8g+8), VGPR4-7 = K[16+8g..16+8g+8), g = lane>>4
__device__ __forceinline__ v16h load_a32(const f16* base, int ld) {
  int lane = threadIdx.x & 31;
  int r = lane & 15, g = lane >> 4;
  const f16* p = base + (size_t)r * ld + (g << 3);
  union { v16h v; v8h h[2]; } u;
  u.h[0] = *(const v8h*)(p);
  u.h[1] = *(const v8h*)(p + 16);
  return u.v;
}
// B (32x16, f16) given B^T row-major: lane n (=lane&15) holds B^T[n, 8g..8g+8) and B^T[n, 16+8g..16+8g+8)
__device__ __forceinline__ v16h load_b32(const f16* base, int ld) {
  int lane = threadIdx.x & 31;
  int r = lane & 15, g = lane >> 4;
  const f16* p = base + (size_t)r * ld + (g << 3);
  union { v16h v; v8h h[2]; } u;
  u.h[0] = *(const v8h*)(p);
  u.h[1] = *(const v8h*)(p + 16);
  return u.v;
}
__device__ __forceinline__ v8f wmma32(v16h a, v16h b, v8f c) {
  return __builtin_amdgcn_wmma_f32_16x16x32_f16(false, a, false, b, (short)0, c, false, false);
}

// ---- elementwise f32 -> f16 ----
__global__ void k_cvt(const float* __restrict__ s, f16* __restrict__ d, int n) {
  int i = blockIdx.x * blockDim.x + threadIdx.x;
  if (i < n) d[i] = (f16)s[i];
}

// ---- QKV projection: qkv = X(8192x1024) @ Wqkv^T(1024x3072) + bias ----
// Per wave: 32(M) x 64(N) tile, double-buffered fragments.
// q,k written f32 in (N,H,L,hd); v written f16 transposed (N,H,hd,L)
__global__ void __launch_bounds__(256) k_gemm_qkv(
    const f16* __restrict__ X, const f16* __restrict__ W,
    const float* __restrict__ bias,
    float* __restrict__ qf, float* __restrict__ kf, f16* __restrict__ vT) {
  const int NT = (3 * CD) / 64; // 48 column tiles of width 64
  int wid = blockIdx.x * (blockDim.x >> 5) + (threadIdx.x >> 5);
  int lane = threadIdx.x & 31;
  int tm = (wid / NT) << 5;     // 32-row tile
  int tn = (wid % NT) << 6;
  const f16* pa0 = X + (size_t)tm * CD;
  const f16* pa1 = X + (size_t)(tm + 16) * CD;
  const f16* pb  = W + (size_t)tn * CD;

  v8f c[2][4] = {};
  v16h a0 = load_a32(pa0, CD);
  v16h a1 = load_a32(pa1, CD);
  v16h b0 = load_b32(pb, CD);
  v16h b1 = load_b32(pb + (size_t)16 * CD, CD);
  v16h b2 = load_b32(pb + (size_t)32 * CD, CD);
  v16h b3 = load_b32(pb + (size_t)48 * CD, CD);
  for (int k0 = 32; k0 <= CD; k0 += 32) {
    int kn = k0 & (CD - 1);     // wraps to 0 on last iteration (harmless reload)
    v16h na0 = load_a32(pa0 + kn, CD);
    v16h na1 = load_a32(pa1 + kn, CD);
    v16h nb0 = load_b32(pb + kn, CD);
    v16h nb1 = load_b32(pb + (size_t)16 * CD + kn, CD);
    v16h nb2 = load_b32(pb + (size_t)32 * CD + kn, CD);
    v16h nb3 = load_b32(pb + (size_t)48 * CD + kn, CD);
    c[0][0] = wmma32(a0, b0, c[0][0]);
    c[0][1] = wmma32(a0, b1, c[0][1]);
    c[0][2] = wmma32(a0, b2, c[0][2]);
    c[0][3] = wmma32(a0, b3, c[0][3]);
    c[1][0] = wmma32(a1, b0, c[1][0]);
    c[1][1] = wmma32(a1, b1, c[1][1]);
    c[1][2] = wmma32(a1, b2, c[1][2]);
    c[1][3] = wmma32(a1, b3, c[1][3]);
    a0 = na0; a1 = na1; b0 = nb0; b1 = nb1; b2 = nb2; b3 = nb3;
  }

  int g = lane >> 4, nl = lane & 15;
#pragma unroll
  for (int mt = 0; mt < 2; ++mt) {
#pragma unroll
    for (int i = 0; i < 8; ++i) {
      int m = tm + (mt << 4) + (g << 3) + i;   // row in (L,N) flatten
      int l = m >> 3, nb = m & 7;
#pragma unroll
      for (int t = 0; t < 4; ++t) {
        int j = tn + (t << 4) + nl;            // column in [0,3C)
        float acc = c[mt][t][i] + bias[j];
        int which = j >> 10;
        int cidx = j & (CD - 1);
        int h = cidx >> 6, d = cidx & 63;
        size_t qk_idx = ((((size_t)nb * NH + h) * LQ + l) << 6) + d;
        if (which == 0)      qf[qk_idx] = acc;
        else if (which == 1) kf[qk_idx] = acc;
        else vT[(((size_t)nb * NH + h) * HD + d) * LQ + l] = (f16)acc;
      }
    }
  }
}

// ---- per-row L2 normalize (hd=64), f32 in -> f16 out; one wave per row ----
__global__ void __launch_bounds__(256) k_norm(
    const float* __restrict__ qf, const float* __restrict__ kf,
    f16* __restrict__ qh, f16* __restrict__ kh) {
  const float* src = blockIdx.y ? kf : qf;
  f16* dst = blockIdx.y ? kh : qh;
  int rowid = blockIdx.x * (blockDim.x >> 5) + (threadIdx.x >> 5);
  int lane = threadIdx.x & 31;
  const float* p = src + ((size_t)rowid << 6) + (lane << 1);
  float a = p[0], b = p[1];
  float ss = a * a + b * b;
  ss += __shfl_xor(ss, 1);
  ss += __shfl_xor(ss, 2);
  ss += __shfl_xor(ss, 4);
  ss += __shfl_xor(ss, 8);
  ss += __shfl_xor(ss, 16);
  float sc = 1.0f / fmaxf(sqrtf(ss), 1e-12f);
  f16* q = dst + ((size_t)rowid << 6) + (lane << 1);
  q[0] = (f16)(a * sc);
  q[1] = (f16)(b * sc);
}

// ---- flash attention: grid (L/128, N*H), 8 waves/block, 16 queries/wave ----
// V-fragment and next-iteration K-fragment loads are issued before the softmax
// VALU section so they overlap it; K-frags rotate across iterations.
__global__ void __launch_bounds__(256) k_attn(
    const f16* __restrict__ qh, const f16* __restrict__ kh, const f16* __restrict__ vT,
    const float* __restrict__ logit_scale, const float* __restrict__ head_scale,
    f16* __restrict__ o) {
  __shared__ f16 ldsP[8][16 * 32];
  int wave = threadIdx.x >> 5;
  int lane = threadIdx.x & 31;
  int nh = blockIdx.y;            // nb*NH + h
  int h = nh & (NH - 1);
  int q0 = blockIdx.x * 128 + (wave << 4);
  const f16* qb = qh + (size_t)nh * LQ * HD;
  const f16* kb = kh + (size_t)nh * LQ * HD;
  const f16* vb = vT + (size_t)nh * HD * LQ;
  float ls = __expf(fminf(logit_scale[h], 4.6051701860f)); // log(1/0.01)
  float hs = head_scale[h];
  int g = lane >> 4, nl = lane & 15;

  v16h aq0 = load_a32(qb + (size_t)q0 * HD, HD);       // dims 0..31
  v16h aq1 = load_a32(qb + (size_t)q0 * HD + 32, HD);  // dims 32..63
  v8f o0{}, o1{}, o2{}, o3{};
  float mrow[8], lrow[8];
#pragma unroll
  for (int i = 0; i < 8; ++i) { mrow[i] = -1e30f; lrow[i] = 0.f; }

  f16* myP = &ldsP[wave][0];
  // preload K fragments for kc = 0
  v16h bk0 = load_b32(kb, HD);
  v16h bk1 = load_b32(kb + 32, HD);
  v16h bk2 = load_b32(kb + (size_t)16 * HD, HD);
  v16h bk3 = load_b32(kb + (size_t)16 * HD + 32, HD);

  for (int kc = 0; kc < LQ; kc += 32) {
    v8f s0{}, s1{};
    s0 = wmma32(aq0, bk0, s0);
    s0 = wmma32(aq1, bk1, s0);
    s1 = wmma32(aq0, bk2, s1);
    s1 = wmma32(aq1, bk3, s1);

    // issue V loads for this chunk (independent of S) — overlap softmax
    v16h bv0 = load_b32(vb + (size_t)0  * LQ + kc, LQ);
    v16h bv1 = load_b32(vb + (size_t)16 * LQ + kc, LQ);
    v16h bv2 = load_b32(vb + (size_t)32 * LQ + kc, LQ);
    v16h bv3 = load_b32(vb + (size_t)48 * LQ + kc, LQ);
    // issue next-iteration K loads (wrap on last iteration)
    int kn = (kc + 32) & (LQ - 1);
    bk0 = load_b32(kb + (size_t)kn * HD, HD);
    bk1 = load_b32(kb + (size_t)kn * HD + 32, HD);
    bk2 = load_b32(kb + (size_t)(kn + 16) * HD, HD);
    bk3 = load_b32(kb + (size_t)(kn + 16) * HD + 32, HD);

#pragma unroll
    for (int i = 0; i < 8; ++i) {
      float a0 = s0[i] * ls, a1 = s1[i] * ls;
      float mx = fmaxf(a0, a1);
      mx = fmaxf(mx, __shfl_xor(mx, 1));
      mx = fmaxf(mx, __shfl_xor(mx, 2));
      mx = fmaxf(mx, __shfl_xor(mx, 4));
      mx = fmaxf(mx, __shfl_xor(mx, 8));
      float mnew = fmaxf(mrow[i], mx);
      float corr = __expf(mrow[i] - mnew);
      float p0 = __expf(a0 - mnew), p1 = __expf(a1 - mnew);
      float rs = p0 + p1;
      rs += __shfl_xor(rs, 1);
      rs += __shfl_xor(rs, 2);
      rs += __shfl_xor(rs, 4);
      rs += __shfl_xor(rs, 8);
      lrow[i] = lrow[i] * corr + rs;
      mrow[i] = mnew;
      o0[i] *= corr; o1[i] *= corr; o2[i] *= corr; o3[i] *= corr;
      int r = (g << 3) + i;                 // C-layout row
      myP[r * 32 + nl] = (f16)p0;           // key col 0..15
      myP[r * 32 + 16 + nl] = (f16)p1;      // key col 16..31
    }
    // transpose P through LDS into A-fragment layout (same-wave DS, in-order)
    union { v16h v; v8h hh[2]; } up;
    const f16* pr = myP + (size_t)nl * 32 + (g << 3);
    up.hh[0] = *(const v8h*)(pr);
    up.hh[1] = *(const v8h*)(pr + 16);
    v16h ap = up.v;
    o0 = wmma32(ap, bv0, o0);
    o1 = wmma32(ap, bv1, o1);
    o2 = wmma32(ap, bv2, o2);
    o3 = wmma32(ap, bv3, o3);
  }
  int nb = nh >> 4;
#pragma unroll
  for (int i = 0; i < 8; ++i) {
    float inv = hs / lrow[i];
    int m = q0 + (g << 3) + i;   // query index l
    size_t base = (((size_t)m * NBATCH + nb) << 10) + (h << 6) + nl; // (L,N,C)
    o[base + 0]  = (f16)(o0[i] * inv);
    o[base + 16] = (f16)(o1[i] * inv);
    o[base + 32] = (f16)(o2[i] * inv);
    o[base + 48] = (f16)(o3[i] * inv);
  }
}

// ---- output projection: out = O(8192x1024) @ out_w^T(1024x1024) + out_b ----
// Per wave: 32(M) x 64(N) tile, double-buffered fragments.
__global__ void __launch_bounds__(256) k_gemm_out(
    const f16* __restrict__ O, const f16* __restrict__ W,
    const float* __restrict__ bias, float* __restrict__ out) {
  const int NT = CD / 64; // 16
  int wid = blockIdx.x * (blockDim.x >> 5) + (threadIdx.x >> 5);
  int lane = threadIdx.x & 31;
  int tm = (wid / NT) << 5;
  int tn = (wid % NT) << 6;
  const f16* pa0 = O + (size_t)tm * CD;
  const f16* pa1 = O + (size_t)(tm + 16) * CD;
  const f16* pb  = W + (size_t)tn * CD;

  v8f c[2][4] = {};
  v16h a0 = load_a32(pa0, CD);
  v16h a1 = load_a32(pa1, CD);
  v16h b0 = load_b32(pb, CD);
  v16h b1 = load_b32(pb + (size_t)16 * CD, CD);
  v16h b2 = load_b32(pb + (size_t)32 * CD, CD);
  v16h b3 = load_b32(pb + (size_t)48 * CD, CD);
  for (int k0 = 32; k0 <= CD; k0 += 32) {
    int kn = k0 & (CD - 1);
    v16h na0 = load_a32(pa0 + kn, CD);
    v16h na1 = load_a32(pa1 + kn, CD);
    v16h nb0 = load_b32(pb + kn, CD);
    v16h nb1 = load_b32(pb + (size_t)16 * CD + kn, CD);
    v16h nb2 = load_b32(pb + (size_t)32 * CD + kn, CD);
    v16h nb3 = load_b32(pb + (size_t)48 * CD + kn, CD);
    c[0][0] = wmma32(a0, b0, c[0][0]);
    c[0][1] = wmma32(a0, b1, c[0][1]);
    c[0][2] = wmma32(a0, b2, c[0][2]);
    c[0][3] = wmma32(a0, b3, c[0][3]);
    c[1][0] = wmma32(a1, b0, c[1][0]);
    c[1][1] = wmma32(a1, b1, c[1][1]);
    c[1][2] = wmma32(a1, b2, c[1][2]);
    c[1][3] = wmma32(a1, b3, c[1][3]);
    a0 = na0; a1 = na1; b0 = nb0; b1 = nb1; b2 = nb2; b3 = nb3;
  }

  int g = lane >> 4, nl = lane & 15;
#pragma unroll
  for (int mt = 0; mt < 2; ++mt) {
#pragma unroll
    for (int i = 0; i < 8; ++i) {
      int m = tm + (mt << 4) + (g << 3) + i;
      size_t rowb = (size_t)m * CD + tn + nl;
#pragma unroll
      for (int t = 0; t < 4; ++t) {
        out[rowb + (t << 4)] = c[mt][t][i] + bias[tn + (t << 4) + nl];
      }
    }
  }
}

extern "C" void kernel_launch(void* const* d_in, const int* in_sizes, int n_in,
                              void* d_out, int out_size, void* d_ws, size_t ws_size,
                              hipStream_t stream) {
  const float* x    = (const float*)d_in[0];   // (L,N,C) = 8M
  const float* wqkv = (const float*)d_in[1];   // (3C,C)  = 3M
  const float* bqkv = (const float*)d_in[2];   // (3C,)
  const float* lsc  = (const float*)d_in[3];   // (H,1,1)
  const float* hsc  = (const float*)d_in[4];   // (H,1,1)
  const float* ow   = (const float*)d_in[5];   // (C,C)
  const float* ob   = (const float*)d_in[6];   // (C,)
  float* out = (float*)d_out;

  char* ws = (char*)d_ws;
  size_t off = 0;
  auto alloc = [&](size_t bytes) -> void* {
    void* p = ws + off;
    off += bytes;
    off = (off + 255) & ~(size_t)255;
    return p;
  };
  const size_t NX = (size_t)MROWS * CD;        // 8388608
  const size_t NW = (size_t)3 * CD * CD;       // 3145728
  const size_t NO = (size_t)CD * CD;           // 1048576
  f16*   Xh  = (f16*)alloc(NX * 2);
  f16*   Wh  = (f16*)alloc(NW * 2);
  f16*   OWh = (f16*)alloc(NO * 2);
  float* qf  = (float*)alloc(NX * 4);
  float* kf  = (float*)alloc(NX * 4);
  f16*   qhh = (f16*)alloc(NX * 2);
  f16*   khh = (f16*)alloc(NX * 2);
  f16*   vT  = (f16*)alloc(NX * 2);
  f16*   Oo  = (f16*)alloc(NX * 2);

  k_cvt<<<(int)((NX + 255) / 256), 256, 0, stream>>>(x, Xh, (int)NX);
  k_cvt<<<(int)((NW + 255) / 256), 256, 0, stream>>>(wqkv, Wh, (int)NW);
  k_cvt<<<(int)((NO + 255) / 256), 256, 0, stream>>>(ow, OWh, (int)NO);

  // 256 row-tiles * 48 col-tiles = 12288 waves / 8 per block
  k_gemm_qkv<<<1536, 256, 0, stream>>>(Xh, Wh, bqkv, qf, kf, vT);

  // 131072 rows per tensor, 8 rows per block, y selects q/k
  k_norm<<<dim3(16384, 2), 256, 0, stream>>>(qf, kf, qhh, khh);

  // (L/128, N*H)
  k_attn<<<dim3(8, 128), 256, 0, stream>>>(qhh, khh, vT, lsc, hsc, Oo);

  // 256 row-tiles * 16 col-tiles = 4096 waves / 8 per block
  k_gemm_out<<<512, 256, 0, stream>>>(Oo, OWh, ob, out);
}
